// BoringFeedForwardMOE_80281528697021
// MI455X (gfx1250) — matrix operand order repaired and checked by reading, not compiled, not run on hardware
//
#include <hip/hip_runtime.h>

// Problem constants (B=4, T=2048 -> N=8192 tokens)
#define NTOK 8192
#define CDIM 1024
#define HDIM 4096
#define EEXP 8
#define TOPK 2
#define CAP  2048   // int(N*K/E * 1.0)

typedef __attribute__((ext_vector_type(16))) __bf16 v16bf;
typedef __attribute__((ext_vector_type(8)))  __bf16 v8bf;
typedef __attribute__((ext_vector_type(4)))  __bf16 v4bf;
typedef __attribute__((ext_vector_type(8)))  float  v8f;
typedef __attribute__((ext_vector_type(4)))  int    v4i;

// global/LDS address-space pointer types for the async-to-LDS builtin
typedef v4i __attribute__((address_space(1)))* gas_v4i_p;
typedef v4i __attribute__((address_space(3)))* las_v4i_p;

__device__ __forceinline__ void async_wait0() {
  __builtin_amdgcn_s_wait_asynccnt(0);
}

// ---------------- utility kernels ----------------

__global__ void zero_f4(float4* __restrict__ p, int n4) {
  int i = blockIdx.x * blockDim.x + threadIdx.x;
  if (i < n4) p[i] = make_float4(0.f, 0.f, 0.f, 0.f);
}

__global__ void cvt_bf16_kernel(const float4* __restrict__ src, v4bf* __restrict__ dst, int n4) {
  int i = blockIdx.x * blockDim.x + threadIdx.x;
  if (i < n4) {
    float4 v = src[i];
    v4bf o;
    o[0] = (__bf16)v.x; o[1] = (__bf16)v.y; o[2] = (__bf16)v.z; o[3] = (__bf16)v.w;
    dst[i] = o;
  }
}

// ---------------- router: logits, top-2, softmax ----------------

__global__ void router_kernel(const float* __restrict__ x, const float* __restrict__ Wr,
                              const float* __restrict__ br, int* __restrict__ topi,
                              float* __restrict__ gates) {
  int t = blockIdx.x * blockDim.x + threadIdx.x;
  if (t >= NTOK) return;
  float acc[EEXP];
#pragma unroll
  for (int e = 0; e < EEXP; ++e) acc[e] = br[e];
  const float* xr = x + (size_t)t * CDIM;
  for (int c = 0; c < CDIM; ++c) {
    float xv = xr[c];
#pragma unroll
    for (int e = 0; e < EEXP; ++e) acc[e] += xv * Wr[c * EEXP + e];
  }
  int i0 = 0; float v0 = acc[0];
#pragma unroll
  for (int e = 1; e < EEXP; ++e) if (acc[e] > v0) { v0 = acc[e]; i0 = e; }
  int i1 = (i0 == 0) ? 1 : 0; float v1 = -3.4e38f;
#pragma unroll
  for (int e = 0; e < EEXP; ++e) if (e != i0 && acc[e] > v1) { v1 = acc[e]; i1 = e; }
  float g0 = 1.0f / (1.0f + __expf(v1 - v0));   // softmax over {v0, v1}
  topi[2 * t + 0] = i0;  topi[2 * t + 1] = i1;
  gates[2 * t + 0] = g0; gates[2 * t + 1] = 1.0f - g0;
}

// ------------- ordered capacity routing (single workgroup scan) -------------

__global__ __launch_bounds__(256)
void scatter_kernel(const int* __restrict__ topi, const float* __restrict__ gates,
                    int* __restrict__ buf, float* __restrict__ wbuf) {
  __shared__ int cnt[256][EEXP];
  const int tid = threadIdx.x;
  for (int idx = tid; idx < EEXP * CAP; idx += 256) { buf[idx] = NTOK; wbuf[idx] = 0.0f; }
  const int CHUNK = NTOK / 256;
  const int t0 = tid * CHUNK;
  int c[EEXP];
#pragma unroll
  for (int e = 0; e < EEXP; ++e) c[e] = 0;
  for (int t = t0; t < t0 + CHUNK; ++t) { c[topi[2 * t]]++; c[topi[2 * t + 1]]++; }
#pragma unroll
  for (int e = 0; e < EEXP; ++e) cnt[tid][e] = c[e];
  __syncthreads();
  if (tid < EEXP) {
    int run = 0;
    for (int i = 0; i < 256; ++i) { int v = cnt[i][tid]; cnt[i][tid] = run; run += v; }
  }
  __syncthreads();
  int r[EEXP];
#pragma unroll
  for (int e = 0; e < EEXP; ++e) r[e] = cnt[tid][e];
  for (int t = t0; t < t0 + CHUNK; ++t) {
#pragma unroll
    for (int k = 0; k < TOPK; ++k) {
      int e = topi[2 * t + k];
      int rk = r[e]++;                       // rank counts even past cap (matches ref)
      if (rk < CAP) { buf[e * CAP + rk] = t; wbuf[e * CAP + rk] = gates[2 * t + k]; }
    }
  }
}

// ---------------- gather tokens into bf16 expert buffers ----------------

__global__ void gather_kernel(const float* __restrict__ x, const int* __restrict__ buf,
                              __bf16* __restrict__ xg) {
  const int row = blockIdx.x;          // 0 .. E*CAP-1
  const int t = buf[row];
  const int c = threadIdx.x * 4;       // 256 threads * 4 = CDIM
  v4bf o;
  if (t < NTOK) {
    const float4 v = *reinterpret_cast<const float4*>(x + (size_t)t * CDIM + c);
    o[0] = (__bf16)v.x; o[1] = (__bf16)v.y; o[2] = (__bf16)v.z; o[3] = (__bf16)v.w;
  } else {
    o[0] = (__bf16)0.f; o[1] = (__bf16)0.f; o[2] = (__bf16)0.f; o[3] = (__bf16)0.f;
  }
  *reinterpret_cast<v4bf*>(xg + (size_t)row * CDIM + c) = o;
}

// ---------------- WMMA GEMM tiles ----------------
// Block: 128x128 output, 256 threads = 8 wave32 waves (2x4 wave grid).
// Wave tile: 64x32 = 4x2 WMMA 16x16 accumulators. K-step 64 = two WMMA K slices
// per barrier window (16 WMMAs/wave between barriers).
// LDS: A row-major [128][72] bf16, staged via GLOBAL_LOAD_ASYNC_TO_LDS_B128
// (ASYNCcnt path, bypasses VGPRs); B transposed [128 n][72 k] staged through
// VGPRs with packed ds_store_b32. The 72-elem (144 B) row stride keeps every
// 16B fragment load aligned and maps rows to distinct bank groups.

#define KSTEP 64
#define LDSTG 72

__device__ __forceinline__ v16bf fragA(const __bf16* lds, int base, int kh8) {
  // A 16x32 layout: lanes 0-15 hold K 0-7 (lo) & 16-23 (hi); lanes 16-31 K 8-15 & 24-31
  v8bf lo = *reinterpret_cast<const v8bf*>(lds + base + kh8);
  v8bf hi = *reinterpret_cast<const v8bf*>(lds + base + 16 + kh8);
  return __builtin_shufflevector(lo, hi, 0,1,2,3,4,5,6,7,8,9,10,11,12,13,14,15);
}

__device__ __forceinline__ v16bf fragB(const __bf16* lds, int base, int kh16) {
  // B 32x16 layout: lanes 0-15 hold K 0-7 (lo) & 16-23 (hi); lanes 16-31 K 8-15 & 24-31
  v8bf lo = *reinterpret_cast<const v8bf*>(lds + base + kh16 / 2);
  v8bf hi = *reinterpret_cast<const v8bf*>(lds + base + 16 + kh16 / 2);
  return __builtin_shufflevector(lo, hi, 0,1,2,3,4,5,6,7,8,9,10,11,12,13,14,15);
}

// Stage A tile 128xKSTEP: direct global->LDS async copy (ASYNCcnt tracked)
__device__ __forceinline__ void stageA(const __bf16* __restrict__ A, __bf16* lds,
                                       int m0, int kk, int Kd, int tid) {
  const int row = tid >> 1, cseg = (tid & 1) * 32;   // 128 rows x 64 cols
  const __bf16* g = A + (size_t)(m0 + row) * Kd + kk + cseg;
  __bf16* l = lds + row * LDSTG + cseg;
  gas_v4i_p gp = (gas_v4i_p)(g);
  las_v4i_p lp = (las_v4i_p)(l);
  __builtin_amdgcn_global_load_async_to_lds_b128(gp, lp, 0, 0);
  __builtin_amdgcn_global_load_async_to_lds_b128(gp, lp, 16, 0);
  __builtin_amdgcn_global_load_async_to_lds_b128(gp, lp, 32, 0);
  __builtin_amdgcn_global_load_async_to_lds_b128(gp, lp, 48, 0);
}

// Stage B tile KSTEPx128 transposed into [n][k]; pack (k,k+1) pairs -> ds_store_b32
__device__ __forceinline__ void stageB(const __bf16* __restrict__ Bm, __bf16* lds,
                                       int n0, int kk, int Nd, int tid) {
  const int kp = (tid & 31) * 2;        // k, k+1 rows (0..62)
  const int nseg = (tid >> 5) * 16;     // 16 columns
  const __bf16* g0 = Bm + (size_t)(kk + kp) * Nd + n0 + nseg;
  const __bf16* g1 = g0 + Nd;
  v8bf r0a = *reinterpret_cast<const v8bf*>(g0);
  v8bf r0b = *reinterpret_cast<const v8bf*>(g0 + 8);
  v8bf r1a = *reinterpret_cast<const v8bf*>(g1);
  v8bf r1b = *reinterpret_cast<const v8bf*>(g1 + 8);
  unsigned* lb = reinterpret_cast<unsigned*>(lds);
  const int kpair = kp >> 1;
#pragma unroll
  for (int j = 0; j < 8; ++j) {
    unsigned lo = (unsigned)__builtin_bit_cast(unsigned short, r0a[j]);
    unsigned hi = (unsigned)__builtin_bit_cast(unsigned short, r1a[j]);
    lb[(nseg + j) * (LDSTG / 2) + kpair] = lo | (hi << 16);
  }
#pragma unroll
  for (int j = 0; j < 8; ++j) {
    unsigned lo = (unsigned)__builtin_bit_cast(unsigned short, r0b[j]);
    unsigned hi = (unsigned)__builtin_bit_cast(unsigned short, r1b[j]);
    lb[(nseg + 8 + j) * (LDSTG / 2) + kpair] = lo | (hi << 16);
  }
}

// GEMM1: h = gelu(xg @ W1 + b1), per expert, M=CAP K=CDIM N=HDIM, bf16 out
__global__ __launch_bounds__(256)
void gemm1_kernel(const __bf16* __restrict__ xg, const __bf16* __restrict__ w1,
                  const float* __restrict__ b1, __bf16* __restrict__ hbuf) {
  __shared__ __align__(16) __bf16 ldsA[128 * LDSTG];
  __shared__ __align__(16) __bf16 ldsB[128 * LDSTG];
  const int e = blockIdx.z;
  const __bf16* A  = xg + (size_t)e * CAP * CDIM;
  const __bf16* Bm = w1 + (size_t)e * CDIM * HDIM;
  const float* bias = b1 + (size_t)e * HDIM;
  __bf16* outp = hbuf + (size_t)e * CAP * HDIM;
  const int m0 = blockIdx.y * 128, n0 = blockIdx.x * 128;
  const int tid = threadIdx.x, lane = tid & 31, wave = tid >> 5;
  const int wm = (wave >> 2) * 64, wn = (wave & 3) * 32;
  v8f acc[4][2];
#pragma unroll
  for (int i = 0; i < 4; ++i)
#pragma unroll
    for (int j = 0; j < 2; ++j) { v8f z = {}; acc[i][j] = z; }
  const int lrow = lane & 15, kh8 = (lane >> 4) * 8, kh16 = (lane >> 4) * 16;
  for (int kk = 0; kk < CDIM; kk += KSTEP) {
    stageA(A, ldsA, m0, kk, CDIM, tid);
    if (kk + KSTEP < CDIM)   // prefetch next B tile rows toward near caches
      __builtin_prefetch(Bm + (size_t)(kk + KSTEP + (tid & 31) * 2) * HDIM + n0, 0, 1);
    stageB(Bm, ldsB, n0, kk, HDIM, tid);
    async_wait0();
    __syncthreads();
#pragma unroll
    for (int ks = 0; ks < KSTEP; ks += 32) {
      v16bf af[4], bfr[2];
#pragma unroll
      for (int i = 0; i < 4; ++i) af[i] = fragA(ldsA, (wm + i * 16 + lrow) * LDSTG + ks, kh8);
#pragma unroll
      for (int j = 0; j < 2; ++j) bfr[j] = fragB(ldsB, (wn + j * 16 + lrow) * LDSTG + ks, kh16);
#pragma unroll
      for (int i = 0; i < 4; ++i)
#pragma unroll
        for (int j = 0; j < 2; ++j)
          acc[i][j] = __builtin_amdgcn_wmma_f32_16x16x32_bf16(
              false, af[i], false, bfr[j], (short)0, acc[i][j], false, false);
    }
    __syncthreads();
  }
  // epilogue: exact GELU, bf16 store (stores coalesce across each half-wave)
  const int ln = lane & 15, mh = (lane >> 4) * 8;
#pragma unroll
  for (int i = 0; i < 4; ++i)
#pragma unroll
    for (int j = 0; j < 2; ++j) {
      int gn = n0 + wn + j * 16 + ln;
      float bv = bias[gn];
#pragma unroll
      for (int r = 0; r < 8; ++r) {
        int gm = m0 + wm + i * 16 + mh + r;
        float v = acc[i][j][r] + bv;
        float gl = 0.5f * v * (1.0f + erff(v * 0.70710678118654752f));
        outp[(size_t)gm * HDIM + gn] = (__bf16)gl;
      }
    }
}

// GEMM2: out[token] += gate * (h @ W2 + b2), per expert, M=CAP K=HDIM N=CDIM
__global__ __launch_bounds__(256)
void gemm2_kernel(const __bf16* __restrict__ hbuf, const __bf16* __restrict__ w2,
                  const float* __restrict__ b2, const int* __restrict__ buf,
                  const float* __restrict__ wbuf, float* __restrict__ outp) {
  __shared__ __align__(16) __bf16 ldsA[128 * LDSTG];
  __shared__ __align__(16) __bf16 ldsB[128 * LDSTG];
  const int e = blockIdx.z;
  const __bf16* A  = hbuf + (size_t)e * CAP * HDIM;
  const __bf16* Bm = w2 + (size_t)e * HDIM * CDIM;
  const float* bias = b2 + (size_t)e * CDIM;
  const int m0 = blockIdx.y * 128, n0 = blockIdx.x * 128;
  const int tid = threadIdx.x, lane = tid & 31, wave = tid >> 5;
  const int wm = (wave >> 2) * 64, wn = (wave & 3) * 32;
  v8f acc[4][2];
#pragma unroll
  for (int i = 0; i < 4; ++i)
#pragma unroll
    for (int j = 0; j < 2; ++j) { v8f z = {}; acc[i][j] = z; }
  const int lrow = lane & 15, kh8 = (lane >> 4) * 8, kh16 = (lane >> 4) * 16;
  for (int kk = 0; kk < HDIM; kk += KSTEP) {
    stageA(A, ldsA, m0, kk, HDIM, tid);
    if (kk + KSTEP < HDIM)
      __builtin_prefetch(Bm + (size_t)(kk + KSTEP + (tid & 31) * 2) * CDIM + n0, 0, 1);
    stageB(Bm, ldsB, n0, kk, CDIM, tid);
    async_wait0();
    __syncthreads();
#pragma unroll
    for (int ks = 0; ks < KSTEP; ks += 32) {
      v16bf af[4], bfr[2];
#pragma unroll
      for (int i = 0; i < 4; ++i) af[i] = fragA(ldsA, (wm + i * 16 + lrow) * LDSTG + ks, kh8);
#pragma unroll
      for (int j = 0; j < 2; ++j) bfr[j] = fragB(ldsB, (wn + j * 16 + lrow) * LDSTG + ks, kh16);
#pragma unroll
      for (int i = 0; i < 4; ++i)
#pragma unroll
        for (int j = 0; j < 2; ++j)
          acc[i][j] = __builtin_amdgcn_wmma_f32_16x16x32_bf16(
              false, af[i], false, bfr[j], (short)0, acc[i][j], false, false);
    }
    __syncthreads();
  }
  // epilogue: bias, gate weight, atomic scatter-add into token rows
  const int ln = lane & 15, mh = (lane >> 4) * 8;
#pragma unroll
  for (int i = 0; i < 4; ++i)
#pragma unroll
    for (int j = 0; j < 2; ++j) {
      int gn = n0 + wn + j * 16 + ln;
      float bv = bias[gn];
#pragma unroll
      for (int r = 0; r < 8; ++r) {
        int gm = m0 + wm + i * 16 + mh + r;
        int tok = buf[e * CAP + gm];
        float w = wbuf[e * CAP + gm];
        float v = (acc[i][j][r] + bv) * w;
        if (tok < NTOK) atomicAdd(outp + (size_t)tok * CDIM + gn, v);
      }
    }
}

// ---------------- host launcher ----------------

extern "C" void kernel_launch(void* const* d_in, const int* in_sizes, int n_in,
                              void* d_out, int out_size, void* d_ws, size_t ws_size,
                              hipStream_t stream) {
  (void)in_sizes; (void)n_in; (void)out_size; (void)ws_size;
  const float* x  = (const float*)d_in[0];
  const float* Wr = (const float*)d_in[1];
  const float* br = (const float*)d_in[2];
  const float* W1 = (const float*)d_in[3];
  const float* b1 = (const float*)d_in[4];
  const float* W2 = (const float*)d_in[5];
  const float* b2 = (const float*)d_in[6];
  float* out = (float*)d_out;

  char* ws = (char*)d_ws;
  size_t off = 0;
  auto take = [&](size_t bytes) -> void* {
    off = (off + 255) & ~(size_t)255;
    void* p = ws + off;
    off += bytes;
    return p;
  };
  int*    topi  = (int*)   take((size_t)NTOK * TOPK * 4);
  float*  gates = (float*) take((size_t)NTOK * TOPK * 4);
  int*    buf   = (int*)   take((size_t)EEXP * CAP * 4);
  float*  wbuf  = (float*) take((size_t)EEXP * CAP * 4);
  __bf16* xg    = (__bf16*)take((size_t)EEXP * CAP * CDIM * 2);
  __bf16* w1b   = (__bf16*)take((size_t)EEXP * CDIM * HDIM * 2);
  __bf16* w2b   = (__bf16*)take((size_t)EEXP * HDIM * CDIM * 2);
  __bf16* hbuf  = (__bf16*)take((size_t)EEXP * CAP * HDIM * 2);

  { int n4 = NTOK * CDIM / 4;
    zero_f4<<<(n4 + 255) / 256, 256, 0, stream>>>((float4*)out, n4); }
  { int n4 = EEXP * CDIM * HDIM / 4;
    cvt_bf16_kernel<<<(n4 + 255) / 256, 256, 0, stream>>>((const float4*)W1, (v4bf*)w1b, n4); }
  { int n4 = EEXP * HDIM * CDIM / 4;
    cvt_bf16_kernel<<<(n4 + 255) / 256, 256, 0, stream>>>((const float4*)W2, (v4bf*)w2b, n4); }

  router_kernel<<<NTOK / 256, 256, 0, stream>>>(x, Wr, br, topi, gates);
  scatter_kernel<<<1, 256, 0, stream>>>(topi, gates, buf, wbuf);
  gather_kernel<<<EEXP * CAP, 256, 0, stream>>>(x, buf, xg);

  gemm1_kernel<<<dim3(HDIM / 128, CAP / 128, EEXP), 256, 0, stream>>>(xg, w1b, b1, hbuf);
  gemm2_kernel<<<dim3(CDIM / 128, CAP / 128, EEXP), 256, 0, stream>>>(hbuf, w2b, b2, buf, wbuf, out);
}
